// MoEExpertPool_39633958208071
// MI455X (gfx1250) — compile-verified, matrix operand order checked by reading, not run on hardware
//
#include <hip/hip_runtime.h>
#include <hip/hip_bf16.h>
#include <math.h>

// Problem constants (match reference)
#define BATCH 4096
#define DIM   2048
#define NEXP  4
#define NGRP  3
#define EPSV  1e-8f

// GEMM tiling
#define BM 128
#define BN 128
#define BK 32
#define LDST 40   // LDS row stride in bf16 elems (80 bytes, multiple of 16B)

typedef __bf16 bf16_t;
typedef bf16_t v16bf __attribute__((ext_vector_type(16)));
typedef bf16_t v8bf  __attribute__((ext_vector_type(8)));
typedef float  v8f   __attribute__((ext_vector_type(8)));
typedef int    v4i_  __attribute__((ext_vector_type(4)));

// ---- async global->LDS path (gfx1250), with safe fallback ----------------
#if defined(__has_builtin)
# if __has_builtin(__builtin_amdgcn_global_load_async_to_lds_b128)
#  define HAVE_ASYNC_LDS 1
# endif
# if __has_builtin(__builtin_amdgcn_s_wait_asynccnt)
#  define HAVE_WAIT_ASYNC 1
# endif
#endif
#ifndef HAVE_ASYNC_LDS
# define HAVE_ASYNC_LDS 0
#endif
#ifndef HAVE_WAIT_ASYNC
# define HAVE_WAIT_ASYNC 0
#endif

#if HAVE_ASYNC_LDS
typedef __attribute__((address_space(1))) v4i_ gv4i;   // global int4
typedef __attribute__((address_space(3))) v4i_ sv4i;   // LDS int4
#endif

__device__ __forceinline__ void async_copy_b128(const bf16_t* g, bf16_t* s) {
#if HAVE_ASYNC_LDS
    __builtin_amdgcn_global_load_async_to_lds_b128(
        (gv4i*)g, (sv4i*)s, 0, 0);
#else
    *(uint4*)s = *(const uint4*)g;
#endif
}

__device__ __forceinline__ void wait_async_lds() {
#if HAVE_ASYNC_LDS
# if HAVE_WAIT_ASYNC
    __builtin_amdgcn_s_wait_asynccnt(0);
# else
    asm volatile("s_wait_asynccnt 0x0" ::: "memory");
# endif
#endif
}

union Frag { v16bf v; v8bf h[2]; };
union PackBf2 { bf16_t h[2]; unsigned int u; };

// Load a 16x32 bf16 WMMA operand fragment from an LDS tile stored with row
// stride LDST. idx = matrix row (A) or column (B, stored transposed).
// Lane layout per CDNA5 ISA: lanes 0-15 hold K in {0..7,16..23}, lanes 16-31
// hold K in {8..15,24..31}.
__device__ __forceinline__ v16bf load_frag(const bf16_t* lds, int idx, int lane) {
    const int c0 = (lane >> 4) << 3;            // 0 or 8
    const bf16_t* p = lds + idx * LDST;
    Frag f;
    f.h[0] = *(const v8bf*)(p + c0);            // K = c0 .. c0+7
    f.h[1] = *(const v8bf*)(p + c0 + 16);       // K = c0+16 .. c0+23
    return f.v;
}

// ---------------------------------------------------------------- utilities
__global__ void convert_bf16_kernel(const float* __restrict__ in,
                                    bf16_t* __restrict__ out, int n) {
    int i = (blockIdx.x * blockDim.x + threadIdx.x) * 4;
    if (i + 3 < n) {
        float4 f = *(const float4*)(in + i);
        out[i + 0] = (bf16_t)f.x;
        out[i + 1] = (bf16_t)f.y;
        out[i + 2] = (bf16_t)f.z;
        out[i + 3] = (bf16_t)f.w;
    }
}

__global__ void zero_f32_kernel(float* __restrict__ p, int n) {
    int i = (blockIdx.x * blockDim.x + threadIdx.x) * 4;
    if (i + 3 < n) {
        float4 z = {0.f, 0.f, 0.f, 0.f};
        *(float4*)(p + i) = z;
    }
}

// ------------------------------------------------------------------- gates
// gates[b][12]: per-group softmax(x@Wg+bg), cb group scaled by (1-mask_mean)
__global__ void gate_kernel(const float* __restrict__ x,
                            const int*   __restrict__ mmask,
                            const float* __restrict__ Wg0, const float* __restrict__ bg0,
                            const float* __restrict__ Wg1, const float* __restrict__ bg1,
                            const float* __restrict__ Wg2, const float* __restrict__ bg2,
                            float* __restrict__ gates) {
    int b = blockIdx.x * blockDim.x + threadIdx.x;
    if (b >= BATCH) return;
    float acc[12];
    for (int e = 0; e < 4; ++e) {
        acc[e]     = bg0[e];
        acc[4 + e] = bg1[e];
        acc[8 + e] = bg2[e];
    }
    const float* xr = x + (size_t)b * DIM;
    for (int k = 0; k < DIM; ++k) {
        float xv = xr[k];
        const float* w0 = Wg0 + k * 4;
        const float* w1 = Wg1 + k * 4;
        const float* w2 = Wg2 + k * 4;
        for (int e = 0; e < 4; ++e) {
            acc[e]     += xv * w0[e];
            acc[4 + e] += xv * w1[e];
            acc[8 + e] += xv * w2[e];
        }
    }
    float mm = (float)(mmask[b * 3] + mmask[b * 3 + 1] + mmask[b * 3 + 2]) * (1.0f / 3.0f);
    float scale[3] = {1.0f, 1.0f - mm, 1.0f};
    for (int g = 0; g < 3; ++g) {
        float mx = acc[4 * g];
        for (int e = 1; e < 4; ++e) mx = fmaxf(mx, acc[4 * g + e]);
        float s = 0.f, ex[4];
        for (int e = 0; e < 4; ++e) { ex[e] = __expf(acc[4 * g + e] - mx); s += ex[e]; }
        float inv = scale[g] / s;
        for (int e = 0; e < 4; ++e) gates[(size_t)b * 12 + 4 * g + e] = ex[e] * inv;
    }
}

// ------------------------------------------------- GEMM 1: h = relu(x@W1+b)
__global__ __launch_bounds__(256) void gemm_relu_kernel(
        const bf16_t* __restrict__ A,     // [BATCH,DIM] bf16
        const float*  __restrict__ W,     // [DIM,DIM]   f32 (k-major rows)
        const float*  __restrict__ bias,  // [DIM]
        bf16_t* __restrict__ H)           // [BATCH,DIM] bf16
{
    __shared__ bf16_t As[BM * LDST];
    __shared__ bf16_t Bs[BN * LDST];

    const int tid  = threadIdx.x;
    const int lane = tid & 31;
    const int wave = tid >> 5;
    const int bm0  = blockIdx.y * BM;
    const int bn0  = blockIdx.x * BN;
    const int wr   = (wave & 3) * 32;   // wave row offset (2 x 16)
    const int wc   = (wave >> 2) * 64;  // wave col offset (4 x 16)

    v8f acc[2][4];
    for (int i = 0; i < 2; ++i)
        for (int j = 0; j < 4; ++j) acc[i][j] = (v8f){0.f,0.f,0.f,0.f,0.f,0.f,0.f,0.f};

    // A staging: per-thread row + 16-col half, moved via async b128 pairs
    const int ar = tid >> 1;            // 0..127
    const int ac = (tid & 1) * 16;      // 0 or 16
    // B staging: per-thread K-pair, 8 output columns (packed b32 stores)
    const int kp = tid & 15;            // k pair index: k = 2*kp, 2*kp+1
    const int n0 = (tid >> 4) * 8;      // 16 groups * 8 = 128 cols

    unsigned int* Bs32 = (unsigned int*)Bs;

    for (int k0 = 0; k0 < DIM; k0 += BK) {
        // W rows 2kp and 2kp+1, 8 fp32 each, into registers
        const float4* g0 = (const float4*)(W + (size_t)(k0 + 2 * kp)     * DIM + bn0 + n0);
        const float4* g1 = (const float4*)(W + (size_t)(k0 + 2 * kp + 1) * DIM + bn0 + n0);
        float4 a0 = g0[0], a1 = g0[1];
        float4 b0 = g1[0], b1 = g1[1];
        if (k0 + BK < DIM)
            __builtin_prefetch(W + (size_t)(k0 + BK + 2 * kp) * DIM + bn0 + n0, 0, 1);

        __syncthreads();                       // prior iteration done reading LDS
        // A tile -> LDS (async DMA path on gfx1250)
        {
            const bf16_t* g = A + (size_t)(bm0 + ar) * DIM + k0 + ac;
            bf16_t* s = As + ar * LDST + ac;
            async_copy_b128(g, s);
            async_copy_b128(g + 8, s + 8);
        }
        // B tile: convert + pack K-pairs, transposed [n][k]
        {
            float lo[8] = {a0.x,a0.y,a0.z,a0.w, a1.x,a1.y,a1.z,a1.w};
            float hi[8] = {b0.x,b0.y,b0.z,b0.w, b1.x,b1.y,b1.z,b1.w};
            #pragma unroll
            for (int i = 0; i < 8; ++i) {
                PackBf2 p;
                p.h[0] = (bf16_t)lo[i];        // k = 2kp
                p.h[1] = (bf16_t)hi[i];        // k = 2kp+1
                Bs32[(n0 + i) * (LDST / 2) + kp] = p.u;
            }
        }
        wait_async_lds();
        __syncthreads();

        v16bf af[2], bf[4];
        #pragma unroll
        for (int i = 0; i < 2; ++i) af[i] = load_frag(As, wr + i * 16 + (lane & 15), lane);
        #pragma unroll
        for (int j = 0; j < 4; ++j) bf[j] = load_frag(Bs, wc + j * 16 + (lane & 15), lane);
        #pragma unroll
        for (int i = 0; i < 2; ++i)
            #pragma unroll
            for (int j = 0; j < 4; ++j)
                acc[i][j] = __builtin_amdgcn_wmma_f32_16x16x32_bf16(
                    false, af[i], false, bf[j], (short)0, acc[i][j], false, false);
    }

    const int lrow = (lane >> 4) * 8;
    const int lcol = lane & 15;
    #pragma unroll
    for (int i = 0; i < 2; ++i)
        #pragma unroll
        for (int j = 0; j < 4; ++j) {
            int colg = bn0 + wc + j * 16 + lcol;
            float bv = bias[colg];
            #pragma unroll
            for (int g = 0; g < 8; ++g) {
                int rowg = bm0 + wr + i * 16 + lrow + g;
                float v = acc[i][j][g] + bv;
                v = v > 0.f ? v : 0.f;
                H[(size_t)rowg * DIM + colg] = (bf16_t)v;
            }
        }
}

// --------------------- GEMM 2: mu/lv + gating + product-of-experts update
__global__ __launch_bounds__(256) void gemm_mulv_kernel(
        const bf16_t* __restrict__ Hn,    // [BATCH,DIM] bf16
        const float*  __restrict__ Wmu, const float* __restrict__ bmu,
        const float*  __restrict__ Wlv, const float* __restrict__ blv,
        const float*  __restrict__ gates, int gidx,
        float* __restrict__ prec_sum, float* __restrict__ mu_prec)
{
    __shared__ bf16_t As [BM * LDST];
    __shared__ bf16_t Bmu[BN * LDST];
    __shared__ bf16_t Blv[BN * LDST];

    const int tid  = threadIdx.x;
    const int lane = tid & 31;
    const int wave = tid >> 5;
    const int bm0  = blockIdx.y * BM;
    const int bn0  = blockIdx.x * BN;
    const int wr   = (wave & 3) * 32;
    const int wc   = (wave >> 2) * 64;

    v8f accM[2][4], accL[2][4];
    for (int i = 0; i < 2; ++i)
        for (int j = 0; j < 4; ++j) {
            accM[i][j] = (v8f){0.f,0.f,0.f,0.f,0.f,0.f,0.f,0.f};
            accL[i][j] = (v8f){0.f,0.f,0.f,0.f,0.f,0.f,0.f,0.f};
        }

    const int ar = tid >> 1;
    const int ac = (tid & 1) * 16;
    const int kp = tid & 15;
    const int n0 = (tid >> 4) * 8;

    unsigned int* Bmu32 = (unsigned int*)Bmu;
    unsigned int* Blv32 = (unsigned int*)Blv;

    for (int k0 = 0; k0 < DIM; k0 += BK) {
        const float4* gm0 = (const float4*)(Wmu + (size_t)(k0 + 2 * kp)     * DIM + bn0 + n0);
        const float4* gm1 = (const float4*)(Wmu + (size_t)(k0 + 2 * kp + 1) * DIM + bn0 + n0);
        const float4* gl0 = (const float4*)(Wlv + (size_t)(k0 + 2 * kp)     * DIM + bn0 + n0);
        const float4* gl1 = (const float4*)(Wlv + (size_t)(k0 + 2 * kp + 1) * DIM + bn0 + n0);
        float4 m00 = gm0[0], m01 = gm0[1], m10 = gm1[0], m11 = gm1[1];
        float4 l00 = gl0[0], l01 = gl0[1], l10 = gl1[0], l11 = gl1[1];
        if (k0 + BK < DIM) {
            __builtin_prefetch(Wmu + (size_t)(k0 + BK + 2 * kp) * DIM + bn0 + n0, 0, 1);
            __builtin_prefetch(Wlv + (size_t)(k0 + BK + 2 * kp) * DIM + bn0 + n0, 0, 1);
        }

        __syncthreads();
        {
            const bf16_t* g = Hn + (size_t)(bm0 + ar) * DIM + k0 + ac;
            bf16_t* s = As + ar * LDST + ac;
            async_copy_b128(g, s);
            async_copy_b128(g + 8, s + 8);
        }
        {
            float mlo[8] = {m00.x,m00.y,m00.z,m00.w, m01.x,m01.y,m01.z,m01.w};
            float mhi[8] = {m10.x,m10.y,m10.z,m10.w, m11.x,m11.y,m11.z,m11.w};
            float llo[8] = {l00.x,l00.y,l00.z,l00.w, l01.x,l01.y,l01.z,l01.w};
            float lhi[8] = {l10.x,l10.y,l10.z,l10.w, l11.x,l11.y,l11.z,l11.w};
            #pragma unroll
            for (int i = 0; i < 8; ++i) {
                PackBf2 pm, pl;
                pm.h[0] = (bf16_t)mlo[i]; pm.h[1] = (bf16_t)mhi[i];
                pl.h[0] = (bf16_t)llo[i]; pl.h[1] = (bf16_t)lhi[i];
                Bmu32[(n0 + i) * (LDST / 2) + kp] = pm.u;
                Blv32[(n0 + i) * (LDST / 2) + kp] = pl.u;
            }
        }
        wait_async_lds();
        __syncthreads();

        v16bf af[2], bm_[4], bl_[4];
        #pragma unroll
        for (int i = 0; i < 2; ++i) af[i] = load_frag(As, wr + i * 16 + (lane & 15), lane);
        #pragma unroll
        for (int j = 0; j < 4; ++j) {
            bm_[j] = load_frag(Bmu, wc + j * 16 + (lane & 15), lane);
            bl_[j] = load_frag(Blv, wc + j * 16 + (lane & 15), lane);
        }
        #pragma unroll
        for (int i = 0; i < 2; ++i)
            #pragma unroll
            for (int j = 0; j < 4; ++j) {
                accM[i][j] = __builtin_amdgcn_wmma_f32_16x16x32_bf16(
                    false, af[i], false, bm_[j], (short)0, accM[i][j], false, false);
                accL[i][j] = __builtin_amdgcn_wmma_f32_16x16x32_bf16(
                    false, af[i], false, bl_[j], (short)0, accL[i][j], false, false);
            }
    }

    const int lrow = (lane >> 4) * 8;
    const int lcol = lane & 15;
    #pragma unroll
    for (int i = 0; i < 2; ++i)
        #pragma unroll
        for (int j = 0; j < 4; ++j) {
            int colg = bn0 + wc + j * 16 + lcol;
            float bmv = bmu[colg];
            float blvv = blv[colg];
            #pragma unroll
            for (int g = 0; g < 8; ++g) {
                int rowg = bm0 + wr + i * 16 + lrow + g;
                float w  = gates[(size_t)rowg * 12 + gidx];
                float mu = accM[i][j][g] + bmv;
                float lv = accL[i][j][g] + blvv;
                float prec = 1.0f / (__expf(lv) + EPSV);
                size_t o = (size_t)rowg * DIM + colg;
                prec_sum[o] += prec;            // tiles disjoint; launches ordered
                mu_prec[o]  += mu * w * prec;
            }
        }
}

// --------------------------------------------------------------- finalize
__global__ void finalize_kernel(const float* __restrict__ prec_sum,
                                const float* __restrict__ mu_prec,
                                float* __restrict__ out, int n) {
    int i = blockIdx.x * blockDim.x + threadIdx.x;
    if (i >= n) return;
    float ps  = prec_sum[i];
    float var = 1.0f / ps;
    out[i]     = mu_prec[i] * var;          // mu_fused
    out[n + i] = __logf(var + EPSV);        // logvar_fused
}

// ---------------------------------------------------------------- launcher
extern "C" void kernel_launch(void* const* d_in, const int* in_sizes, int n_in,
                              void* d_out, int out_size, void* d_ws, size_t ws_size,
                              hipStream_t stream) {
    (void)in_sizes; (void)n_in; (void)out_size; (void)ws_size;
    const float* x     = (const float*)d_in[0];
    const int*   mmask = (const int*)d_in[1];
    const float* grp[NGRP][8];
    for (int g = 0; g < NGRP; ++g)
        for (int j = 0; j < 8; ++j)
            grp[g][j] = (const float*)d_in[2 + g * 8 + j];
    // group param order: W1, b1, Wmu, bmu, Wlv, blv, Wg, bg

    // workspace carve-up (~101 MB total)
    const size_t N = (size_t)BATCH * DIM;
    bf16_t* xbf      = (bf16_t*)d_ws;
    bf16_t* hbf      = xbf + N;
    float*  prec_sum = (float*)(hbf + N);
    float*  mu_prec  = prec_sum + N;
    float*  gates    = mu_prec + N;

    // 1. x -> bf16
    convert_bf16_kernel<<<(int)(N / (256 * 4)), 256, 0, stream>>>(x, xbf, (int)N);
    // 2. zero PoE accumulators
    zero_f32_kernel<<<(int)(N / (256 * 4)), 256, 0, stream>>>(prec_sum, (int)N);
    zero_f32_kernel<<<(int)(N / (256 * 4)), 256, 0, stream>>>(mu_prec, (int)N);
    // 3. gates (all three groups, mask folded into cb)
    gate_kernel<<<BATCH / 256, 256, 0, stream>>>(
        x, mmask, grp[0][6], grp[0][7], grp[1][6], grp[1][7], grp[2][6], grp[2][7], gates);

    // 4. 12 experts: h-GEMM then mu/lv-GEMM with PoE accumulation
    dim3 gg(DIM / BN, BATCH / BM);   // (16, 32)
    for (int g = 0; g < NGRP; ++g) {
        for (int e = 0; e < NEXP; ++e) {
            const size_t wo = (size_t)e * DIM * DIM;
            const size_t bo = (size_t)e * DIM;
            gemm_relu_kernel<<<gg, 256, 0, stream>>>(
                xbf, grp[g][0] + wo, grp[g][1] + bo, hbf);
            gemm_mulv_kernel<<<gg, 256, 0, stream>>>(
                hbf,
                grp[g][2] + wo, grp[g][3] + bo,
                grp[g][4] + wo, grp[g][5] + bo,
                gates, g * NEXP + e, prec_sum, mu_prec);
        }
    }

    // 5. fuse
    finalize_kernel<<<(int)((N + 255) / 256), 256, 0, stream>>>(
        prec_sum, mu_prec, (float*)d_out, (int)N);
}
